// TensorNet_5325759447733
// MI455X (gfx1250) — compile-verified, matrix operand order checked by reading, not run on hardware
//
#include <hip/hip_runtime.h>
#include <hip/hip_bf16.h>
#include <math.h>

#define HC 128          // hidden channels
#define RC 32           // num_rbf
#define LC 2            // layers
#define CUTOFF_F 4.5f

typedef __attribute__((ext_vector_type(16))) _Float16 v16h;
typedef __attribute__((ext_vector_type(8)))  float    v8f;

union Frag16 { v16h v; uint4 q[2]; };

// 16-bit WMMA operand fragment (ISA 7.12.2): lane groups 0-15/16-31 split K by 8;
// slots 0..7 = K[kb+sel*8 ..], slots 8..15 = K[kb+16+sel*8 ..]  -> two 128-bit loads.
__device__ __forceinline__ v16h load_frag_g(const _Float16* __restrict__ base,
                                            int row, int stride, int kb, int sel) {
    const _Float16* p = base + (size_t)row * stride + kb + sel * 8;
    Frag16 f;
    f.q[0] = *reinterpret_cast<const uint4*>(p);
    f.q[1] = *reinterpret_cast<const uint4*>(p + 16);
    return f.v;
}

__device__ __forceinline__ v16h load_frag_lds(const _Float16* base,
                                              int row, int stride, int kb, int sel) {
    const _Float16* p = base + row * stride + kb + sel * 8;
    Frag16 f;
    f.q[0] = *reinterpret_cast<const uint4*>(p);   // ds_load_b128
    f.q[1] = *reinterpret_cast<const uint4*>(p + 16);
    return f.v;
}

// ---------------- Unified WMMA GEMM ----------------
// out[M,N] = act(A[M,K] @ W[N,K]^T + bias) * rowscale
// Block mapping: blockIdx.x = group + nGroups * blkInGroup, group = 64-column slice.
// Weight slice (64 x K, f16, padded rows K+8) staged once in LDS; each wave computes
// 16x64 output (4 accumulators) reusing one A fragment per K-step.
// planeMode: 9 planar GEMMs on blockIdx.z (iso|A|S), weight from Wt[0/1/2], f32 out.
__global__ void wmma_gemm(const _Float16* __restrict__ A,
                          const _Float16* __restrict__ W,
                          const float* __restrict__ bias,
                          const float* __restrict__ rowscale,
                          _Float16* __restrict__ outH,
                          float* __restrict__ outF,
                          int M, int N, int K, int act, int planeMode) {
    extern __shared__ _Float16 smem[];
    const int LK = K + 8;                     // padded LDS row stride (halves)

    const _Float16* Ap = A;
    const _Float16* Wp = W;
    float* outFp = outF;
    if (planeMode) {
        const int p = blockIdx.z;
        const int widx = (p == 0) ? 0 : (p < 4 ? 1 : 2);
        Ap    += (size_t)p * M * K;
        Wp    += (size_t)widx * N * K;
        outFp += (size_t)p * M * N;
    }

    const int nGroups = N >> 6;               // 64-column groups
    const int group   = blockIdx.x % nGroups;
    const int blkInG  = blockIdx.x / nGroups;
    const int blksPerG = gridDim.x / nGroups;

    // Cooperative stage: W rows [group*64, group*64+64) -> LDS, padded rows.
    {
        const _Float16* Wg = Wp + (size_t)group * 64 * K;
        const int vecPerRow = K >> 3;         // uint4 per row
        for (int v = threadIdx.x; v < 64 * vecPerRow; v += blockDim.x) {
            const int r = v / vecPerRow;
            const int o = v - r * vecPerRow;
            *reinterpret_cast<uint4*>(smem + r * LK + o * 8) =
                *reinterpret_cast<const uint4*>(Wg + (size_t)r * K + o * 8);
        }
    }
    __syncthreads();

    const int waveId = threadIdx.x >> 5;
    const int wavesPerBlock = blockDim.x >> 5;
    const int lane = threadIdx.x & 31;
    const int lm   = lane & 15;
    const int sel  = lane >> 4;
    const int mtiles = M >> 4;

    for (int mt = blkInG * wavesPerBlock + waveId; mt < mtiles;
         mt += blksPerG * wavesPerBlock) {
        const int arow = mt * 16 + lm;
        v8f c0 = {}, c1 = {}, c2 = {}, c3 = {};
        for (int kb = 0; kb < K; kb += 32) {
            v16h a  = load_frag_g(Ap, arow, K, kb, sel);
            v16h b0 = load_frag_lds(smem, 0 * 16 + lm, LK, kb, sel);
            v16h b1 = load_frag_lds(smem, 1 * 16 + lm, LK, kb, sel);
            v16h b2 = load_frag_lds(smem, 2 * 16 + lm, LK, kb, sel);
            v16h b3 = load_frag_lds(smem, 3 * 16 + lm, LK, kb, sel);
            c0 = __builtin_amdgcn_wmma_f32_16x16x32_f16(false, a, false, b0, (short)0, c0, false, false);
            c1 = __builtin_amdgcn_wmma_f32_16x16x32_f16(false, a, false, b1, (short)0, c1, false, false);
            c2 = __builtin_amdgcn_wmma_f32_16x16x32_f16(false, a, false, b2, (short)0, c2, false, false);
            c3 = __builtin_amdgcn_wmma_f32_16x16x32_f16(false, a, false, b3, (short)0, c3, false, false);
        }
        // D layout: VGPR r -> (row = mt*16 + sel*8 + r, col = lane&15 within tile)
        const int rbase = mt * 16 + sel * 8;
        v8f acc[4] = {c0, c1, c2, c3};
        #pragma unroll
        for (int j = 0; j < 4; ++j) {
            const int col = group * 64 + j * 16 + lm;
            const float bv = bias ? bias[col] : 0.0f;
            #pragma unroll
            for (int r = 0; r < 8; ++r) {
                float v = acc[j][r] + bv;
                if (act) v = v / (1.0f + __expf(-v));       // SiLU
                if (rowscale) v *= rowscale[rbase + r];     // cosine cutoff per edge row
                if (outH) outH[(size_t)(rbase + r) * N + col] = (_Float16)v;
                else      outFp[(size_t)(rbase + r) * N + col] = v;
            }
        }
    }
}

// ---------------- Elementwise helpers ----------------
__global__ void cvt_f32_f16(const float* __restrict__ s, _Float16* __restrict__ d, int n) {
    int i = blockIdx.x * blockDim.x + threadIdx.x;
    if (i < n) d[i] = (_Float16)s[i];
}

__global__ void cutoff_kernel(const float* __restrict__ w, float* __restrict__ C, int E) {
    int i = blockIdx.x * blockDim.x + threadIdx.x;
    if (i < E) {
        float x = w[i];
        float c = 0.5f * (__cosf(x * ((float)M_PI / CUTOFF_F)) + 1.0f);
        C[i] = (x < CUTOFF_F) ? c : 0.0f;
    }
}

// Normalize X by (||X||_F^2 + 1), decompose into 9 compressed planes (f32 + f16).
__global__ void decompose_kernel(const float* __restrict__ X,
                                 float* __restrict__ cF, _Float16* __restrict__ cH,
                                 int NHt) {
    int i = blockIdx.x * blockDim.x + threadIdx.x;
    if (i >= NHt) return;
    const float* t = X + (size_t)i * 9;
    float x[9], nrm = 0.0f;
    #pragma unroll
    for (int j = 0; j < 9; ++j) { x[j] = t[j]; nrm += x[j] * x[j]; }
    float inv = 1.0f / (nrm + 1.0f);
    #pragma unroll
    for (int j = 0; j < 9; ++j) x[j] *= inv;
    float iso = (x[0] + x[4] + x[8]) * (1.0f / 3.0f);
    float c[9];
    c[0] = iso;
    c[1] = 0.5f * (x[1] - x[3]);  c[2] = 0.5f * (x[2] - x[6]);  c[3] = 0.5f * (x[5] - x[7]);
    c[4] = x[0] - iso;            c[5] = x[4] - iso;
    c[6] = 0.5f * (x[1] + x[3]);  c[7] = 0.5f * (x[2] + x[6]);  c[8] = 0.5f * (x[5] + x[7]);
    #pragma unroll
    for (int p = 0; p < 9; ++p) {
        cF[(size_t)p * NHt + i] = c[p];
        cH[(size_t)p * NHt + i] = (_Float16)c[p];
    }
}

// Per (edge, h): gather mixed Y planes at dst, weight by ea, atomic-scatter to Mcomp[src].
__global__ void msg_scatter(const _Float16* __restrict__ ea,   // [E, HC, 3]
                            const int* __restrict__ eidx,      // [2, E]
                            const float* __restrict__ Yc,      // [9][N*HC]
                            float* __restrict__ Mc,            // [9][N*HC]
                            int E, int NHt) {
    int t = blockIdx.x * blockDim.x + threadIdx.x;
    if (t >= E * HC) return;
    int e = t / HC, h = t - e * HC;
    int src = eidx[e], dst = eidx[E + e];
    const _Float16* p = ea + (size_t)e * 3 * HC + h * 3;
    float w0 = (float)p[0], w1 = (float)p[1], w2 = (float)p[2];
    int bd = dst * HC + h, bs = src * HC + h;
    atomicAdd(&Mc[bs], w0 * Yc[bd]);
    #pragma unroll
    for (int c = 1; c < 4; ++c)
        atomicAdd(&Mc[(size_t)c * NHt + bs], w1 * Yc[(size_t)c * NHt + bd]);
    #pragma unroll
    for (int c = 4; c < 9; ++c)
        atomicAdd(&Mc[(size_t)c * NHt + bs], w2 * Yc[(size_t)c * NHt + bd]);
}

__device__ __forceinline__ void recon3x3(const float c[9], float T[9]) {
    float iso = c[0], a01 = c[1], a02 = c[2], a12 = c[3];
    float s00 = c[4], s11 = c[5], s01 = c[6], s02 = c[7], s12 = c[8];
    T[0] = iso + s00; T[4] = iso + s11; T[8] = iso - s00 - s11;
    T[1] = s01 + a01; T[3] = s01 - a01;
    T[2] = s02 + a02; T[6] = s02 - a02;
    T[5] = s12 + a12; T[7] = s12 - a12;
}

__device__ __forceinline__ void mm3(const float A[9], const float B[9], float C[9]) {
    #pragma unroll
    for (int i = 0; i < 3; ++i)
        #pragma unroll
        for (int j = 0; j < 3; ++j)
            C[i * 3 + j] = A[i * 3] * B[j] + A[i * 3 + 1] * B[3 + j] + A[i * 3 + 2] * B[6 + j];
}

// AB = M@Y + Y@M, scale by (1+0.1q), decompose, normalize by (||.||^2+1) -> f16 planes
__global__ void tensor_update(const float* __restrict__ Mc, const float* __restrict__ Yc,
                              const float* __restrict__ q, _Float16* __restrict__ c2H,
                              int NHt) {
    int i = blockIdx.x * blockDim.x + threadIdx.x;
    if (i >= NHt) return;
    int n = i / HC;
    float mc[9], yc[9];
    #pragma unroll
    for (int p = 0; p < 9; ++p) { mc[p] = Mc[(size_t)p * NHt + i]; yc[p] = Yc[(size_t)p * NHt + i]; }
    float M[9], Y[9], MY[9], YM[9];
    recon3x3(mc, M); recon3x3(yc, Y);
    mm3(M, Y, MY); mm3(Y, M, YM);
    float s = 1.0f + 0.1f * q[n];
    float T[9], fro = 0.0f;
    #pragma unroll
    for (int j = 0; j < 9; ++j) { T[j] = s * (MY[j] + YM[j]); fro += T[j] * T[j]; }
    float inv = 1.0f / (fro + 1.0f);
    float iso = (T[0] + T[4] + T[8]) * (1.0f / 3.0f);
    float c[9];
    c[0] = iso;
    c[1] = 0.5f * (T[1] - T[3]);  c[2] = 0.5f * (T[2] - T[6]);  c[3] = 0.5f * (T[5] - T[7]);
    c[4] = T[0] - iso;            c[5] = T[4] - iso;
    c[6] = 0.5f * (T[1] + T[3]);  c[7] = 0.5f * (T[2] + T[6]);  c[8] = 0.5f * (T[5] + T[7]);
    #pragma unroll
    for (int p = 0; p < 9; ++p) c2H[(size_t)p * NHt + i] = (_Float16)(c[p] * inv);
}

// out = Xn + dX + (1+0.1q) * dX@dX   (Xn, dX reconstructed from planes)
__global__ void output_kernel(const float* __restrict__ cX, const float* __restrict__ dXc,
                              const float* __restrict__ q, float* __restrict__ out,
                              int NHt) {
    int i = blockIdx.x * blockDim.x + threadIdx.x;
    if (i >= NHt) return;
    int n = i / HC;
    float a[9], b[9];
    #pragma unroll
    for (int p = 0; p < 9; ++p) { a[p] = cX[(size_t)p * NHt + i]; b[p] = dXc[(size_t)p * NHt + i]; }
    float Xn[9], dX[9], P[9];
    recon3x3(a, Xn); recon3x3(b, dX);
    mm3(dX, dX, P);
    float s = 1.0f + 0.1f * q[n];
    float* o = out + (size_t)i * 9;
    #pragma unroll
    for (int j = 0; j < 9; ++j) o[j] = Xn[j] + dX[j] + s * P[j];
}

// ---------------- Host launcher ----------------
extern "C" void kernel_launch(void* const* d_in, const int* in_sizes, int n_in,
                              void* d_out, int out_size, void* d_ws, size_t ws_size,
                              hipStream_t stream) {
    const float* X0   = (const float*)d_in[0];
    const float* attr = (const float*)d_in[1];
    const float* ew   = (const float*)d_in[2];
    const float* q    = (const float*)d_in[3];
    const float* Ws1  = (const float*)d_in[4];
    const float* bs1  = (const float*)d_in[5];
    const float* Ws2  = (const float*)d_in[6];
    const float* bs2  = (const float*)d_in[7];
    const float* Ws3  = (const float*)d_in[8];
    const float* bs3  = (const float*)d_in[9];
    const float* Wt   = (const float*)d_in[10];
    const int*   eidx = (const int*)d_in[11];

    const int E   = in_sizes[2];
    const int N   = in_sizes[3];
    const int NHt = N * HC;

    char* ws = (char*)d_ws;
    size_t off = 0;
    auto alloc = [&](size_t bytes) -> void* {
        void* p = ws + off;
        off = (off + bytes + 255) & ~(size_t)255;
        return p;
    };
    _Float16* attrH = (_Float16*)alloc((size_t)E * RC * 2);
    float*    Cbuf  = (float*)   alloc((size_t)E * 4);
    _Float16* w1H   = (_Float16*)alloc((size_t)HC * RC * 2);
    _Float16* w2H   = (_Float16*)alloc((size_t)2 * HC * HC * 2);
    _Float16* w3H   = (_Float16*)alloc((size_t)6 * HC * HC * 2);
    _Float16* wtH   = (_Float16*)alloc((size_t)6 * HC * HC * 2);
    _Float16* ea1H  = (_Float16*)alloc((size_t)E * HC * 2);
    _Float16* ea2H  = (_Float16*)alloc((size_t)E * 2 * HC * 2);
    _Float16* ea3H  = (_Float16*)alloc((size_t)E * 3 * HC * 2);
    float*    compF = (float*)   alloc((size_t)9 * NHt * 4);
    _Float16* compH = (_Float16*)alloc((size_t)9 * NHt * 2);
    float*    YcF   = (float*)   alloc((size_t)9 * NHt * 4);
    float*    McF   = (float*)   alloc((size_t)9 * NHt * 4);
    _Float16* c2H   = (_Float16*)alloc((size_t)9 * NHt * 2);
    float*    dXF   = (float*)   alloc((size_t)9 * NHt * 4);
    float*    Xbuf  = (float*)   alloc((size_t)NHt * 9 * 4);
    (void)ws_size; (void)n_in; (void)out_size;

    const int TPB = 256;
    const int WPB = TPB / 32;   // waves per block

    // Generic GEMM launch: grid = nGroups * blocksPerGroup (x) * planes (z)
    auto launch_gemm = [&](const _Float16* A, const _Float16* W, const float* bias,
                           const float* rowscale, _Float16* outH, float* outF,
                           int M, int Nn, int K, int act, int planeMode, int planes) {
        const int nGroups = Nn / 64;
        const int mtiles  = M / 16;
        const int blksPerG = (mtiles + WPB - 1) / WPB;
        dim3 grid(nGroups * blksPerG, 1, planes);
        const size_t smemBytes = (size_t)64 * (K + 8) * sizeof(_Float16);
        wmma_gemm<<<grid, TPB, smemBytes, stream>>>(A, W, bias, rowscale, outH, outF,
                                                    M, Nn, K, act, planeMode);
    };

    cvt_f32_f16<<<(E * RC + TPB - 1) / TPB, TPB, 0, stream>>>(attr, attrH, E * RC);
    cutoff_kernel<<<(E + TPB - 1) / TPB, TPB, 0, stream>>>(ew, Cbuf, E);

    for (int l = 0; l < LC; ++l) {
        const float* Xin  = (l == 0) ? X0 : Xbuf;
        float*       Xout = (l == LC - 1) ? (float*)d_out : Xbuf;

        cvt_f32_f16<<<(HC * RC + TPB - 1) / TPB, TPB, 0, stream>>>(Ws1 + (size_t)l * HC * RC, w1H, HC * RC);
        cvt_f32_f16<<<(2 * HC * HC + TPB - 1) / TPB, TPB, 0, stream>>>(Ws2 + (size_t)l * 2 * HC * HC, w2H, 2 * HC * HC);
        cvt_f32_f16<<<(6 * HC * HC + TPB - 1) / TPB, TPB, 0, stream>>>(Ws3 + (size_t)l * 6 * HC * HC, w3H, 6 * HC * HC);
        cvt_f32_f16<<<(6 * HC * HC + TPB - 1) / TPB, TPB, 0, stream>>>(Wt  + (size_t)l * 6 * HC * HC, wtH, 6 * HC * HC);

        // Edge scalar MLP: R -> H -> 2H -> 3H with SiLU; final stage fused with cutoff
        launch_gemm(attrH, w1H, bs1 + (size_t)l * HC,     nullptr, ea1H, nullptr, E, HC,     RC,     1, 0, 1);
        launch_gemm(ea1H,  w2H, bs2 + (size_t)l * 2 * HC, nullptr, ea2H, nullptr, E, 2 * HC, HC,     1, 0, 1);
        launch_gemm(ea2H,  w3H, bs3 + (size_t)l * 3 * HC, Cbuf,    ea3H, nullptr, E, 3 * HC, 2 * HC, 1, 0, 1);

        // Node tensors: normalize + decompose into 9 planes
        decompose_kernel<<<(NHt + TPB - 1) / TPB, TPB, 0, stream>>>(Xin, compF, compH, NHt);

        // Chan-mix set 1 (Wt[0..2]) -> Y planes (9 planar GEMMs on z)
        launch_gemm(compH, wtH, nullptr, nullptr, nullptr, YcF, N, HC, HC, 0, 1, 9);

        // Edge messages + scatter
        hipMemsetAsync(McF, 0, (size_t)9 * NHt * 4, stream);
        msg_scatter<<<(E * HC + TPB - 1) / TPB, TPB, 0, stream>>>(ea3H, eidx, YcF, McF, E, NHt);

        // AB = MY+YM, scale, decompose, normalize
        tensor_update<<<(NHt + TPB - 1) / TPB, TPB, 0, stream>>>(McF, YcF, q, c2H, NHt);

        // Chan-mix set 2 (Wt[3..5]) -> dX planes
        launch_gemm(c2H, wtH + (size_t)3 * HC * HC, nullptr, nullptr, nullptr, dXF, N, HC, HC, 0, 1, 9);

        // out = Xn + dX + (1+0.1q) * dX@dX
        output_kernel<<<(NHt + TPB - 1) / TPB, TPB, 0, stream>>>(compF, dXF, q, Xout, NHt);
    }
}
